// SimpleAttention_44315472560818
// MI455X (gfx1250) — compile-verified, matrix operand order checked by reading, not run on hardware
//
#include <hip/hip_runtime.h>
#include <hip/hip_bf16.h>
#include <math.h>
#include <stdint.h>

typedef __attribute__((ext_vector_type(16))) __bf16 v16bf;
typedef __attribute__((ext_vector_type(8)))  float  v8f;
typedef __attribute__((ext_vector_type(4)))  float  v4f;
typedef __attribute__((ext_vector_type(4)))  unsigned int v4u;
typedef __attribute__((ext_vector_type(4)))  int    v4i;
typedef unsigned int   u32;
typedef unsigned short u16;

#define NN   4096
#define KK   8192
#define SCALE_INV (1.0f / 90.0f)   // SCALE = float(int(sqrt(8192))) = 90

__device__ __forceinline__ u32 bf16_rne(float f) {
    u32 u = __float_as_uint(f);
    return ((u + 0x7FFFu + ((u >> 16) & 1u)) >> 16) & 0xFFFFu;
}
__device__ __forceinline__ u32 pack_bf2(float lo, float hi) {
    return bf16_rne(lo) | (bf16_rne(hi) << 16);
}

union Frag { v16bf v; v4u q[2]; };

// ---- async global->LDS copy (gfx1250 GLOBAL_LOAD_ASYNC_TO_LDS_B128), with
// ---- a plain-store fallback so the file always compiles. -------------------
#if defined(__gfx1250__) &&                                                    \
    __has_builtin(__builtin_amdgcn_global_load_async_to_lds_b128) &&           \
    __has_builtin(__builtin_amdgcn_s_wait_asynccnt)
#define USE_ASYNC_LDS 1
#else
#define USE_ASYNC_LDS 0
#endif

typedef v4i __attribute__((address_space(1)))* gptr_v4i;   // global int4*
typedef v4i __attribute__((address_space(3)))* lptr_v4i;   // LDS    int4*

__device__ __forceinline__ void copy16_g2l(void* lds_dst, const void* gsrc) {
#if USE_ASYNC_LDS
    // generic LDS address truncates to the 32-bit LDS offset (ISA aperture rule);
    // generic global address == addrspace(1) address.
    __builtin_amdgcn_global_load_async_to_lds_b128(
        (gptr_v4i)(uintptr_t)gsrc,
        (lptr_v4i)(u32)(uintptr_t)lds_dst,
        /*offset=*/0, /*cpol=*/0);
#else
    *(v4u*)lds_dst = *(const v4u*)gsrc;
#endif
}
__device__ __forceinline__ void wait_async_copies() {
#if USE_ASYNC_LDS
    __builtin_amdgcn_s_wait_asynccnt(0);
#endif
}

// ---------------------------------------------------------------------------
// Kernel 1: K = X@Wk + bk, V = X@Wv + bv.
// Output tile per WG: 128(M) x 64(N). 8 waves in 4x2 grid, each wave 32x32
// (2x2 WMMA tiles) of BOTH K and V, sharing the X tile in LDS. BK = 32.
// grid = (N/64, N/128), block = 256
// ---------------------------------------------------------------------------
__global__ __launch_bounds__(256)
void gemm_kv_kernel(const float* __restrict__ X,
                    const float* __restrict__ Wk, const float* __restrict__ bk,
                    const float* __restrict__ Wv, const float* __restrict__ bv,
                    u16* __restrict__ Kbf, u16* __restrict__ Vbf,
                    float* __restrict__ Vf)
{
    __shared__ alignas(16) u32 sA [128][16];  // 128 rows x 32 bf16 (64B pitch)
    __shared__ alignas(16) u16 sBk[64][32];   // transposed: [n][k]
    __shared__ alignas(16) u16 sBv[64][32];

    const int tid   = threadIdx.x;
    const int lane  = tid & 31;
    const int wave  = tid >> 5;
    const int lr    = lane & 15;
    const int l16   = lane >> 4;
    const int waveM = wave & 3;   // 4 bands of 32 rows
    const int waveN = wave >> 2;  // 2 bands of 32 cols

    const int n0 = blockIdx.x * 64;
    const int m0 = blockIdx.y * 128;

    v8f accK[2][2] = {};
    v8f accV[2][2] = {};

    for (int kt = 0; kt < KK; kt += 32) {
        // ---- stage A: X[m0..+127][kt..+31] -> bf16 in LDS ----
#pragma unroll
        for (int i = 0; i < 4; ++i) {
            int id = tid + i * 256;
            int row = id >> 3, c4 = id & 7;
            const float* src = X + (size_t)(m0 + row) * KK + kt + c4 * 4;
            v4f x = *(const v4f*)src;
            sA[row][c4 * 2 + 0] = pack_bf2(x[0], x[1]);
            sA[row][c4 * 2 + 1] = pack_bf2(x[2], x[3]);
            if (kt + 32 < KK) __builtin_prefetch((const void*)(src + 32), 0, 1);
        }
        // ---- stage Bk / Bv transposed: sB[n][k] ----
#pragma unroll
        for (int i = 0; i < 2; ++i) {
            int id = tid + i * 256;
            int kk = id >> 4, n4 = id & 15;
            v4f wk = *(const v4f*)(Wk + (size_t)(kt + kk) * NN + n0 + n4 * 4);
            v4f wv = *(const v4f*)(Wv + (size_t)(kt + kk) * NN + n0 + n4 * 4);
#pragma unroll
            for (int j = 0; j < 4; ++j) {
                sBk[n4 * 4 + j][kk] = (u16)bf16_rne(wk[j]);
                sBv[n4 * 4 + j][kk] = (u16)bf16_rne(wv[j]);
            }
        }
        __syncthreads();

        // ---- fragments (ISA 16-bit A/B layout: 2x b128 per fragment) ----
        Frag fa[2], fbk[2], fbv[2];
#pragma unroll
        for (int tm = 0; tm < 2; ++tm) {
            const v4u* rp = (const v4u*)&sA[waveM * 32 + tm * 16 + lr][0];
            fa[tm].q[0] = rp[l16];
            fa[tm].q[1] = rp[l16 + 2];
        }
#pragma unroll
        for (int tn = 0; tn < 2; ++tn) {
            const v4u* rk = (const v4u*)&sBk[waveN * 32 + tn * 16 + lr][0];
            fbk[tn].q[0] = rk[l16]; fbk[tn].q[1] = rk[l16 + 2];
            const v4u* rv = (const v4u*)&sBv[waveN * 32 + tn * 16 + lr][0];
            fbv[tn].q[0] = rv[l16]; fbv[tn].q[1] = rv[l16 + 2];
        }
#pragma unroll
        for (int tm = 0; tm < 2; ++tm)
#pragma unroll
            for (int tn = 0; tn < 2; ++tn) {
                accK[tm][tn] = __builtin_amdgcn_wmma_f32_16x16x32_bf16(
                    false, fa[tm].v, false, fbk[tn].v, (short)0, accK[tm][tn], false, false);
                accV[tm][tn] = __builtin_amdgcn_wmma_f32_16x16x32_bf16(
                    false, fa[tm].v, false, fbv[tn].v, (short)0, accV[tm][tn], false, false);
            }
        __syncthreads();
    }

    // ---- epilogue: add bias, store K/V bf16 and V f32 ----
#pragma unroll
    for (int tm = 0; tm < 2; ++tm)
#pragma unroll
        for (int tn = 0; tn < 2; ++tn)
#pragma unroll
            for (int r = 0; r < 8; ++r) {
                int gm = m0 + waveM * 32 + tm * 16 + r + 8 * l16;   // C layout: M = r + 8*(lane>=16)
                int gn = n0 + waveN * 32 + tn * 16 + lr;            //           N = lane & 15
                float kv = accK[tm][tn][r] + bk[gn];
                float vv = accV[tm][tn][r] + bv[gn];
                size_t off = (size_t)gm * NN + gn;
                Kbf[off] = (u16)bf16_rne(kv);
                Vbf[off] = (u16)bf16_rne(vv);
                Vf[off]  = vv;
            }
}

// ---------------------------------------------------------------------------
// Kernel 2: fused scores -> online softmax -> att*V -> row max/argmax.
// Q == V (reference bug). WG owns 32 rows; 8 waves in 2x4 grid; each wave owns
// a 16(M) x 32(N) band = TWO 16x16 WMMA tiles per k-step. j-tile BN = 128.
// Staging uses async global->LDS when available. grid = N/32, block = 256
// ---------------------------------------------------------------------------
__global__ __launch_bounds__(256)
void attn_kernel(const u16* __restrict__ Qbf, const u16* __restrict__ Kbf,
                 const float* __restrict__ Vf,
                 float* __restrict__ outVals, int* __restrict__ outIdx)
{
    __shared__ alignas(16) u16 sQ[32][32];
    __shared__ alignas(16) u16 sK[128][32];
    __shared__ float runMax[32], runSum[32], runBest[32], mNewSh[32];
    __shared__ int   runIdx[32];
    __shared__ float pmax[32][4], psum[32][4], pbest[32][4];
    __shared__ int   pbidx[32][4];

    const int tid  = threadIdx.x;
    const int lane = tid & 31, wave = tid >> 5;
    const int lr   = lane & 15, l16 = lane >> 4;
    const int tm   = wave & 1;   // row half  (2 x 16 rows)
    const int tn   = wave >> 1;  // 0..3 -> 32-col band (2 subtiles of 16)
    const int i0   = blockIdx.x * 32;

    if (tid < 32) {
        runMax[tid]  = -INFINITY;
        runSum[tid]  = 0.0f;
        runBest[tid] = -INFINITY;
        runIdx[tid]  = 0;
    }
    __syncthreads();

    for (int j0 = 0; j0 < NN; j0 += 128) {
        // ---- S tile [32 x 128] = (V-rows) x (K-rows)^T : accumulate over k ----
        v8f acc0 = {}, acc1 = {};
        for (int kt = 0; kt < NN; kt += 32) {
            if (tid < 128) {      // sQ: 32x32 bf16 = 128 x b128
                int row = tid >> 2, c8 = tid & 3;
                copy16_g2l(&sQ[row][c8 * 8],
                           Qbf + (size_t)(i0 + row) * NN + kt + c8 * 8);
            }
#pragma unroll
            for (int i = 0; i < 2; ++i) {  // sK: 128x32 bf16 = 512 x b128
                int id = tid + i * 256;
                int row = id >> 2, c8 = id & 3;
                copy16_g2l(&sK[row][c8 * 8],
                           Kbf + (size_t)(j0 + row) * NN + kt + c8 * 8);
            }
            wait_async_copies();
            __syncthreads();

            Frag fq, fk0, fk1;
            const v4u* rq = (const v4u*)&sQ[tm * 16 + lr][0];
            fq.q[0] = rq[l16]; fq.q[1] = rq[l16 + 2];
            const v4u* r0 = (const v4u*)&sK[tn * 32 + lr][0];
            fk0.q[0] = r0[l16]; fk0.q[1] = r0[l16 + 2];
            const v4u* r1 = (const v4u*)&sK[tn * 32 + 16 + lr][0];
            fk1.q[0] = r1[l16]; fk1.q[1] = r1[l16 + 2];
            acc0 = __builtin_amdgcn_wmma_f32_16x16x32_bf16(
                false, fq.v, false, fk0.v, (short)0, acc0, false, false);
            acc1 = __builtin_amdgcn_wmma_f32_16x16x32_bf16(
                false, fq.v, false, fk1.v, (short)0, acc1, false, false);
            __syncthreads();
        }

        // ---- scores + per-row tile max (merge both subtiles, then 16-lane reduce) ----
        float s0[8], s1[8];
#pragma unroll
        for (int r = 0; r < 8; ++r) { s0[r] = acc0[r] * SCALE_INV; s1[r] = acc1[r] * SCALE_INV; }
#pragma unroll
        for (int r = 0; r < 8; ++r) {
            float v = fmaxf(s0[r], s1[r]);
#pragma unroll
            for (int off = 1; off < 16; off <<= 1)
                v = fmaxf(v, __shfl_xor(v, off, 32));
            if (lr == 0) pmax[tm * 16 + 8 * l16 + r][tn] = v;
        }
        __syncthreads();

        // ---- update running max; rescale running sum / best ----
        if (tid < 32) {
            float mt = fmaxf(fmaxf(pmax[tid][0], pmax[tid][1]),
                             fmaxf(pmax[tid][2], pmax[tid][3]));
            float mold = runMax[tid];
            float mnew = fmaxf(mold, mt);
            float corr = (mold == -INFINITY) ? 0.0f : expf(mold - mnew);
            runSum[tid] *= corr;
            float ob = runBest[tid];
            runBest[tid] = (ob == -INFINITY) ? -INFINITY : ob * corr;
            runMax[tid] = mnew;
            mNewSh[tid] = mnew;
        }
        __syncthreads();

        // ---- e = exp(s - m); p = e * V[i,j]; reduce sum(e), max(p)+idx ----
#pragma unroll
        for (int r = 0; r < 8; ++r) {
            int row = tm * 16 + 8 * l16 + r;
            float m = mNewSh[row];
            int gj0 = j0 + tn * 32 + lr;
            int gj1 = gj0 + 16;
            const float* vrow = Vf + (size_t)(i0 + row) * NN;
            float e0 = expf(s0[r] - m), e1 = expf(s1[r] - m);
            float p0 = e0 * vrow[gj0], p1 = e1 * vrow[gj1];
            float es = e0 + e1;
            float p  = p0; int bi = gj0;
            if (p1 > p) { p = p1; bi = gj1; }
#pragma unroll
            for (int off = 1; off < 16; off <<= 1) {
                es += __shfl_xor(es, off, 32);
                float op = __shfl_xor(p, off, 32);
                int   oi = __shfl_xor(bi, off, 32);
                if (op > p) { p = op; bi = oi; }
            }
            if (lr == 0) {
                psum [row][tn] = es;
                pbest[row][tn] = p;
                pbidx[row][tn] = bi;
            }
        }
        __syncthreads();

        // ---- merge 4 col-bands into running state ----
        if (tid < 32) {
            runSum[tid] += psum[tid][0] + psum[tid][1] + psum[tid][2] + psum[tid][3];
            float b = runBest[tid]; int bix = runIdx[tid];
#pragma unroll
            for (int t = 0; t < 4; ++t)
                if (pbest[tid][t] > b) { b = pbest[tid][t]; bix = pbidx[tid][t]; }
            runBest[tid] = b; runIdx[tid] = bix;
        }
        __syncthreads();
    }

    if (tid < 32) {
        outVals[i0 + tid] = runBest[tid] / runSum[tid];
        outIdx [i0 + tid] = runIdx[tid];
    }
}

// ---------------------------------------------------------------------------
extern "C" void kernel_launch(void* const* d_in, const int* in_sizes, int n_in,
                              void* d_out, int out_size, void* d_ws, size_t ws_size,
                              hipStream_t stream) {
    (void)in_sizes; (void)n_in; (void)out_size; (void)ws_size;
    const float* X  = (const float*)d_in[0];
    const float* Wk = (const float*)d_in[1];
    const float* bk = (const float*)d_in[2];
    const float* Wv = (const float*)d_in[3];
    const float* bv = (const float*)d_in[4];
    // d_in[5] (Wq), d_in[6] (bq) intentionally unused: reference computes x_q = x@Wv + bv

    // workspace: Kbf (32MB) | Vbf (32MB) | Vf (64MB)  == 128MB
    u16*   Kbf = (u16*)d_ws;
    u16*   Vbf = (u16*)((char*)d_ws + (size_t)NN * NN * 2);
    float* Vf  = (float*)((char*)d_ws + (size_t)NN * NN * 4);

    float* outVals = (float*)d_out;
    int*   outIdx  = (int*)d_out + NN;

    dim3 g1(NN / 64, NN / 128);
    gemm_kv_kernel<<<g1, 256, 0, stream>>>(X, Wk, bk, Wv, bv, Kbf, Vbf, Vf);
    attn_kernel<<<NN / 32, 256, 0, stream>>>(Vbf, Kbf, Vf, outVals, outIdx);
}
